// ConvLayer_2972117369018
// MI455X (gfx1250) — matrix operand order repaired and checked by reading, not compiled, not run on hardware
//
#include <hip/hip_runtime.h>
#include <hip/hip_bf16.h>

typedef __attribute__((ext_vector_type(16))) __bf16 v16bf;
typedef __attribute__((ext_vector_type(8)))  __bf16 v8bf;
typedef __attribute__((ext_vector_type(8)))  float  v8f;

#define H_A   128
#define H_B   64
#define KTOT  320          // 2*H_A + H_B
#define COUT  256          // 2*H_A
#define NN    512
#define MM    32
#define PAIRS 4            // (b,n) pairs per block
#define ROWS  128          // rows per block = PAIRS * 32
#define LDA   328          // 320 + 8 pad (bf16) -> row stride 656B (16B aligned)
#define LDB   40           // 32 + 8 pad        -> row stride 80B  (16B aligned)
#define NBLK  1024         // (8*512)/PAIRS
#define NROWS_TOT  131072  // B*N*M
#define NPAIR_ROWS 4096    // B*N
#define EPS   1e-5f

__device__ __forceinline__ unsigned short f2bf(float f) {
    unsigned u = __float_as_uint(f);
    u += 0x7FFFu + ((u >> 16) & 1u);         // round-to-nearest-even
    return (unsigned short)(u >> 16);
}

__device__ __forceinline__ v16bf frag_cat(const unsigned short* lo, const unsigned short* hi) {
    v8bf a = *(const v8bf*)lo;               // 16B contiguous
    v8bf b = *(const v8bf*)hi;
    return __builtin_shufflevector(a, b, 0,1,2,3,4,5,6,7,8,9,10,11,12,13,14,15);
}

// one-time: W f32 -> bf16 (removes conversions + halves L2 traffic in GEMM passes)
__global__ __launch_bounds__(256)
void wconv_kernel(const float* __restrict__ W, unsigned short* __restrict__ Wbf) {
    int e = blockIdx.x * 256 + threadIdx.x;   // 256*320 = 81920 elems
    Wbf[e] = f2bf(W[e]);
}

// PASS 0: GEMM -> per-channel partial (sum, sumsq) of raw accumulators (bias cancels in BN)
// PASS 1: GEMM -> BN-h affine -> sigmoid(filter)*relu(core) -> sum over M -> nbr_sumed
template<int PASS>
__global__ __launch_bounds__(256)
void conv_gemm_pass(const float* __restrict__ atom_emb,
                    const float* __restrict__ nbr_emb,
                    const long long* __restrict__ adj,
                    const float* __restrict__ amask,
                    const unsigned short* __restrict__ Wbf,
                    float* __restrict__ hPartial,      // pass 0 out: [NBLK][512]
                    const float* __restrict__ hScale,  // pass 1 in:  [256]
                    const float* __restrict__ hShift,  // pass 1 in:  [256]
                    float* __restrict__ nbrSumed)      // pass 1 out: [4096][128]
{
    __shared__ unsigned short sA[ROWS][LDA];     // 83,968 B
    __shared__ unsigned short sB[2][COUT][LDB];  // 40,960 B (double-buffered W slab)
    __shared__ float sRed[512];                  //  2,048 B  (~127 KB total, 2 blocks/WGP)

    const int t     = threadIdx.x;
    const int blk   = blockIdx.x;
    const int wave  = t >> 5;
    const int lane  = t & 31;
    const int hlane = lane & 15;
    const int khalf = lane >> 4;

    // prefetch first W k-slab into independent registers (64 B per thread)
    uint4 w0, w1, w2, w3;
    {
        const uint4* src = (const uint4*)(Wbf + (size_t)t * KTOT);
        w0 = src[0]; w1 = src[1]; w2 = src[2]; w3 = src[3];
    }

    // ---------- build A tile: 128 rows x 320 (gather + concat + mask), bf16 ----------
    // 128*320 f32 = 10240 float4 chunks; 256 threads * 40 chunks; 4-deep MLP batches
    #pragma unroll
    for (int itb = 0; itb < 10; ++itb) {
        float4 v[4];
        int rr[4], kk[4];
        #pragma unroll
        for (int j = 0; j < 4; ++j) {
            int ch = t + (itb * 4 + j) * 256;
            int r  = ch / 80;            // 80 float4 per row
            int k0 = (ch % 80) * 4;
            rr[j] = r; kk[j] = k0;
            int p  = r >> 5;
            int m  = r & 31;
            int bn = blk * PAIRS + p;    // flattened b*512+n
            if (k0 < 128) {
                v[j] = *(const float4*)(atom_emb + (size_t)bn * H_A + k0);
            } else if (k0 < 256) {
                long long jj = adj[(size_t)bn * MM + m];
                float msk   = amask[(size_t)bn * MM + m];
                float4 g = *(const float4*)(atom_emb +
                            ((size_t)(bn >> 9) * NN + (size_t)jj) * H_A + (k0 - 128));
                v[j] = make_float4(g.x * msk, g.y * msk, g.z * msk, g.w * msk);
            } else {
                v[j] = *(const float4*)(nbr_emb + ((size_t)bn * MM + m) * H_B + (k0 - 256));
            }
        }
        #pragma unroll
        for (int j = 0; j < 4; ++j) {
            unsigned short* d = &sA[rr[j]][kk[j]];
            d[0] = f2bf(v[j].x); d[1] = f2bf(v[j].y);
            d[2] = f2bf(v[j].z); d[3] = f2bf(v[j].w);
        }
    }

    // commit slab 0 (its loads completed long ago, during the A build)
    {
        uint4* dst = (uint4*)&sB[0][t][0];
        dst[0] = w0; dst[1] = w1; dst[2] = w2; dst[3] = w3;
    }

    // wave owns all 8 M-tiles x {filter-tile w, core-tile w+8}
    v8f acc[8][2];
    #pragma unroll
    for (int mt = 0; mt < 8; ++mt) { v8f z = {}; acc[mt][0] = z; acc[mt][1] = z; }

    const int koffA = khalf * 8;   // A frag: lanes 0-15 take K 0-7/16-23, lanes 16-31 take 8-15/24-31

    __syncthreads();               // sA + sB[0] visible

    #pragma unroll
    for (int ks = 0; ks < 10; ++ks) {
        const int cur = ks & 1, nxt = cur ^ 1;

        // kick off next slab's global loads; latency overlaps the WMMAs below
        if (ks < 9) {
            const uint4* src = (const uint4*)(Wbf + (size_t)t * KTOT + (ks + 1) * 32);
            w0 = src[0]; w1 = src[1]; w2 = src[2]; w3 = src[3];
        }

        v16bf bF[2];
        #pragma unroll
        for (int s = 0; s < 2; ++s) {
            const unsigned short* pr = &sB[cur][wave * 16 + hlane + s * 128][khalf * 8];
            bF[s] = frag_cat(pr, pr + 16);
        }
        #pragma unroll
        for (int mt = 0; mt < 8; ++mt) {
            const unsigned short* pr = &sA[mt * 16 + hlane][ks * 32 + koffA];
            v16bf aF = frag_cat(pr, pr + 16);
            acc[mt][0] = __builtin_amdgcn_wmma_f32_16x16x32_bf16(
                false, aF, false, bF[0], (short)0, acc[mt][0], false, false);
            acc[mt][1] = __builtin_amdgcn_wmma_f32_16x16x32_bf16(
                false, aF, false, bF[1], (short)0, acc[mt][1], false, false);
        }

        // store next slab into the other buffer — same region as compute, no extra barrier
        if (ks < 9) {
            uint4* dst = (uint4*)&sB[nxt][t][0];
            dst[0] = w0; dst[1] = w1; dst[2] = w2; dst[3] = w3;
            __syncthreads();       // one barrier per K-step
        }
    }

    if (PASS == 0) {
        // per-channel Sx, Sxx over the 128 rows of this block (deterministic, no atomics)
        #pragma unroll
        for (int s = 0; s < 2; ++s) {
            float sum = 0.f, sq = 0.f;
            #pragma unroll
            for (int mt = 0; mt < 8; ++mt)
                #pragma unroll
                for (int i = 0; i < 8; ++i) {
                    float x = acc[mt][s][i];
                    sum += x; sq += x * x;
                }
            sum += __shfl_xor(sum, 16, 32);   // combine row halves (lane <-> lane+16)
            sq  += __shfl_xor(sq,  16, 32);
            if (lane < 16) {
                int c = wave * 16 + hlane + s * 128;
                sRed[c] = sum; sRed[256 + c] = sq;
            }
        }
        __syncthreads();
        hPartial[(size_t)blk * 512 + t]       = sRed[t];
        hPartial[(size_t)blk * 512 + 256 + t] = sRed[256 + t];
    } else {
        const int cf = wave * 16 + hlane;
        const float scF = hScale[cf],       shF = hShift[cf];
        const float scC = hScale[cf + 128], shC = hShift[cf + 128];
        #pragma unroll
        for (int p = 0; p < PAIRS; ++p) {   // p = which (b,n) pair in this block
            float sum = 0.f;
            #pragma unroll
            for (int mt = 2 * p; mt < 2 * p + 2; ++mt)
                #pragma unroll
                for (int i = 0; i < 8; ++i) {
                    float hf = acc[mt][0][i] * scF + shF;
                    float hc = acc[mt][1][i] * scC + shC;
                    float sg = 1.f / (1.f + __expf(-hf));
                    sum += (hc > 0.f) ? sg * hc : 0.f;
                }
            sum += __shfl_xor(sum, 16, 32);  // other 16 rows of this pair
            if (lane < 16) sRed[p * 128 + cf] = sum;
        }
        __syncthreads();
        #pragma unroll
        for (int e = t; e < 512; e += 256) {
            int p = e >> 7, c = e & 127;
            nbrSumed[(size_t)(blk * PAIRS + p) * H_A + c] = sRed[e];
        }
    }
}

__global__ __launch_bounds__(256)
void hstats_kernel(const float* __restrict__ hPartial,
                   const float* __restrict__ gamma_h, const float* __restrict__ beta_h,
                   float* __restrict__ hScale, float* __restrict__ hShift)
{
    int c = threadIdx.x;          // 256 channels
    float s = 0.f, q = 0.f;
    for (int i = 0; i < NBLK; ++i) {
        s += hPartial[(size_t)i * 512 + c];
        q += hPartial[(size_t)i * 512 + 256 + c];
    }
    float invN = 1.f / (float)NROWS_TOT;
    float mean = s * invN;
    float var  = q * invN - mean * mean;
    float sc   = gamma_h[c] * rsqrtf(var + EPS);
    hScale[c] = sc;
    hShift[c] = beta_h[c] - mean * sc;    // bias term cancels exactly in BN
}

__global__ __launch_bounds__(256)
void ostats_kernel(const float* __restrict__ nbrSumed,
                   const float* __restrict__ gamma_o, const float* __restrict__ beta_o,
                   float* __restrict__ oScale, float* __restrict__ oShift)
{
    __shared__ float rs[256], rq[256];
    int c = blockIdx.x;           // 128 channels
    float s = 0.f, q = 0.f;
    for (int r = threadIdx.x; r < NPAIR_ROWS; r += 256) {
        float v = nbrSumed[(size_t)r * H_A + c];
        s += v; q += v * v;
    }
    rs[threadIdx.x] = s; rq[threadIdx.x] = q;
    __syncthreads();
    for (int o = 128; o > 0; o >>= 1) {
        if (threadIdx.x < o) { rs[threadIdx.x] += rs[threadIdx.x + o];
                               rq[threadIdx.x] += rq[threadIdx.x + o]; }
        __syncthreads();
    }
    if (threadIdx.x == 0) {
        float invN = 1.f / (float)NPAIR_ROWS;
        float mean = rs[0] * invN;
        float var  = rq[0] * invN - mean * mean;
        float sc   = gamma_o[c] * rsqrtf(var + EPS);
        oScale[c] = sc;
        oShift[c] = beta_o[c] - mean * sc;
    }
}

__global__ __launch_bounds__(256)
void final_kernel(const float* __restrict__ atom_emb, const float* __restrict__ nbrSumed,
                  const float* __restrict__ oScale, const float* __restrict__ oShift,
                  float* __restrict__ out)
{
    int e = blockIdx.x * 256 + threadIdx.x;   // 8*512*128 = 524288 elems
    int c = e & 127;
    float v = atom_emb[e] + nbrSumed[e] * oScale[c] + oShift[c];
    out[e] = v > 0.f ? v : 0.f;
}

extern "C" void kernel_launch(void* const* d_in, const int* in_sizes, int n_in,
                              void* d_out, int out_size, void* d_ws, size_t ws_size,
                              hipStream_t stream) {
    const float*     atom_emb = (const float*)d_in[0];
    const float*     nbr_emb  = (const float*)d_in[1];
    const long long* adj      = (const long long*)d_in[2];
    const float*     amask    = (const float*)d_in[3];
    const float*     W        = (const float*)d_in[4];
    // d_in[5] = bias b: mathematically cancels inside training-mode BatchNorm
    const float*     gamma_h  = (const float*)d_in[6];
    const float*     beta_h   = (const float*)d_in[7];
    const float*     gamma_o  = (const float*)d_in[8];
    const float*     beta_o   = (const float*)d_in[9];
    float*           out      = (float*)d_out;

    unsigned short* Wbf  = (unsigned short*)d_ws;            // 256*320 bf16 = 163,840 B
    float* ws       = (float*)((char*)d_ws + 163840);
    float* hPartial = ws;                                    // 1024*512 = 524,288 f
    float* hScale   = hPartial + (size_t)NBLK * 512;
    float* hShift   = hScale + 256;
    float* nbrSumed = hShift + 256;                          // 4096*128 = 524,288 f
    float* oScale   = nbrSumed + (size_t)NPAIR_ROWS * H_A;
    float* oShift   = oScale + 128;                          // total ~4.4 MB

    wconv_kernel<<<320, 256, 0, stream>>>(W, Wbf);
    conv_gemm_pass<0><<<NBLK, 256, 0, stream>>>(atom_emb, nbr_emb, adj, amask, Wbf,
                                                hPartial, nullptr, nullptr, nullptr);
    hstats_kernel<<<1, 256, 0, stream>>>(hPartial, gamma_h, beta_h, hScale, hShift);
    conv_gemm_pass<1><<<NBLK, 256, 0, stream>>>(atom_emb, nbr_emb, adj, amask, Wbf,
                                                nullptr, hScale, hShift, nbrSumed);
    ostats_kernel<<<128, 256, 0, stream>>>(nbrSumed, gamma_o, beta_o, oScale, oShift);
    final_kernel<<<2048, 256, 0, stream>>>(atom_emb, nbrSumed, oScale, oShift, out);
}
